// GraphSAGE_38165079392458
// MI455X (gfx1250) — matrix operand order repaired and checked by reading, not compiled, not run on hardware
//
#include <hip/hip_runtime.h>
#include <hip/hip_bf16.h>

typedef __attribute__((ext_vector_type(16))) _Float16 v16h;
typedef __attribute__((ext_vector_type(8)))  float    v8f;

#define F_DIM 64      // F_IN == H == 64 in this model
#define WROW  72      // LDS row stride (halves): 64 + 8 pad -> spreads banks

// ---------------------------------------------------------------------------
// Convert an f32 weight matrix [rows x 64] (row-major) to f16, zero-padding
// rows up to padRows (layer 3: 40 -> 48 so every WMMA B-tile row is valid).
// ---------------------------------------------------------------------------
__global__ void sage_wcvt_kernel(const float* __restrict__ W,
                                 _Float16* __restrict__ Wh,
                                 int rows, int padRows) {
    int t = blockIdx.x * blockDim.x + threadIdx.x;
    int total = padRows * F_DIM;
    if (t >= total) return;
    int r = t >> 6;
    Wh[t] = (r < rows) ? (_Float16)W[t] : (_Float16)0.0f;
}

// Zero-padded f32 bias copy (layer 3: 40 -> 48).
__global__ void sage_bcvt_kernel(const float* __restrict__ b,
                                 float* __restrict__ bp, int dim, int padDim) {
    int t = blockIdx.x * blockDim.x + threadIdx.x;
    if (t >= padDim) return;
    bp[t] = (t < dim) ? b[t] : 0.0f;
}

// ---------------------------------------------------------------------------
// In-degree (float) via global f32 atomics. Computed once, reused all layers.
// ---------------------------------------------------------------------------
__global__ void sage_degree_kernel(const int* __restrict__ dst,
                                   float* __restrict__ deg, int E) {
    int e = blockIdx.x * blockDim.x + threadIdx.x;
    if (e < E) atomicAdd(&deg[dst[e]], 1.0f);
}

// ---------------------------------------------------------------------------
// Scatter-add: agg[dst] += h[src]. One thread per (edge, 4-feature group):
// float4 gather (largely L2-resident: the whole 25.6MB feature matrix fits
// the 192MB L2) + 4 global_atomic_add_f32.
// ---------------------------------------------------------------------------
__global__ void sage_scatter_kernel(const float* __restrict__ hin,
                                    const int* __restrict__ src,
                                    const int* __restrict__ dst,
                                    float* __restrict__ agg, int E) {
    int t = blockIdx.x * blockDim.x + threadIdx.x;
    int e = t >> 4;
    if (e >= E) return;
    int g = (t & 15) << 2;
    int s = src[e];
    int d = dst[e];
    const float4 v = *reinterpret_cast<const float4*>(hin + s * F_DIM + g);
    float* o = agg + d * F_DIM + g;
    atomicAdd(o + 0, v.x);
    atomicAdd(o + 1, v.y);
    atomicAdd(o + 2, v.z);
    atomicAdd(o + 3, v.w);
}

// ---------------------------------------------------------------------------
// A-fragment loader: 16x32 f16 A-tile, CDNA5 layout.
// lane L: row m = L&15, kb = (L>=16) ? 8 : 0
//   halves 0..7  <- row[kb .. kb+7]      (f32 -> f16, scaled)
//   halves 8..15 <- row[kb+16 .. kb+23]
// p points at row + kt*32 + kb (16B aligned -> two b128 load pairs).
// ---------------------------------------------------------------------------
__device__ __forceinline__ v16h load_a_frag(const float* __restrict__ p,
                                            float scale) {
    union { float4 v; float f[4]; } f0, f1, f2, f3;
    f0.v = *reinterpret_cast<const float4*>(p + 0);
    f1.v = *reinterpret_cast<const float4*>(p + 4);
    f2.v = *reinterpret_cast<const float4*>(p + 16);
    f3.v = *reinterpret_cast<const float4*>(p + 20);
    v16h a;
#pragma unroll
    for (int i = 0; i < 4; ++i) {
        a[i]      = (_Float16)(f0.f[i] * scale);
        a[4 + i]  = (_Float16)(f1.f[i] * scale);
        a[8 + i]  = (_Float16)(f2.f[i] * scale);
        a[12 + i] = (_Float16)(f3.f[i] * scale);
    }
    return a;
}

// ---------------------------------------------------------------------------
// B-fragment loader from LDS-staged weights: 32x16 f16 B-tile = W^T tile.
// lane L holds column n = L&15 == row n of row-major W; kb = (L>=16)?8:0,
// halves 0..7 <- K kb..kb+7, halves 8..15 <- K kb+16..kb+23 in the kt chunk.
// wrow = &sW[n*WROW]; offsets stay 16B aligned -> ds_load_b128 x2.
// ---------------------------------------------------------------------------
__device__ __forceinline__ v16h load_b_frag(const _Float16* wrow, int kt,
                                            bool hi) {
    const _Float16* p = wrow + kt * 32 + (hi ? 8 : 0);
    union { uint4 u; _Float16 h[8]; } c0, c1;
    c0.u = *reinterpret_cast<const uint4*>(p);
    c1.u = *reinterpret_cast<const uint4*>(p + 16);
    v16h b;
#pragma unroll
    for (int i = 0; i < 8; ++i) { b[i] = c0.h[i]; b[i + 8] = c1.h[i]; }
    return b;
}

// ---------------------------------------------------------------------------
// out = relu?( (agg/deg) @ Wl^T + bias + hin @ Wr^T ), one 16-node row block
// per wave32, K = 64 (2 chunks of 32), NT output tiles of 16 columns.
// Weights staged in LDS per block; two independent accumulator chains so the
// four WMMAs per tile pipeline. EXEC is all-ones at every WMMA.
// ---------------------------------------------------------------------------
template <int NT>
__global__ __launch_bounds__(256) void sage_gemm_kernel(
    const float* __restrict__ agg, const float* __restrict__ hin,
    const float* __restrict__ deg, const _Float16* __restrict__ Wl,
    const _Float16* __restrict__ Wr, const float* __restrict__ bias,
    float* __restrict__ out, int outDim, int outStride, int relu, int nWaves) {
    __shared__ _Float16 sWl[NT * 16 * WROW];
    __shared__ _Float16 sWr[NT * 16 * WROW];

    int wid  = (blockIdx.x * blockDim.x + threadIdx.x) >> 5;
    int lane = threadIdx.x & 31;

    // Cooperative LDS staging of both weight matrices (b128 copies).
    constexpr int ROWS = NT * 16;
    for (int i = threadIdx.x; i < ROWS * 8; i += 256) {
        int r = i >> 3;
        int c = (i & 7) << 3;  // 8-half (16B) chunks
        *reinterpret_cast<uint4*>(&sWl[r * WROW + c]) =
            *reinterpret_cast<const uint4*>(&Wl[r * F_DIM + c]);
        *reinterpret_cast<uint4*>(&sWr[r * WROW + c]) =
            *reinterpret_cast<const uint4*>(&Wr[r * F_DIM + c]);
    }
    __syncthreads();

    if (wid >= nWaves) return;  // wave-uniform, after barrier participation

    int  nodeBase = wid << 4;
    int  lh  = lane & 15;
    bool hi  = lane >= 16;
    int  kb  = hi ? 8 : 0;
    int  node = nodeBase + lh;  // A row handled by this lane

    float inv = 1.0f / fmaxf(deg[node], 1.0f);  // mean aggregation scale
    const float* arow = agg + node * F_DIM;
    const float* hrow = hin + node * F_DIM;

    // Preload all four A fragments (agg/deg and hin, K chunks 0 and 1).
    v16h aAgg0 = load_a_frag(arow + 0 * 32 + kb, inv);
    v16h aAgg1 = load_a_frag(arow + 1 * 32 + kb, inv);
    v16h aH0   = load_a_frag(hrow + 0 * 32 + kb, 1.0f);
    v16h aH1   = load_a_frag(hrow + 1 * 32 + kb, 1.0f);

#pragma unroll
    for (int nt = 0; nt < NT; ++nt) {
        int   n  = (nt << 4) + lh;   // D column for this lane
        float bv = bias[n];          // padded: always valid

        const _Float16* wlrow = &sWl[n * WROW];
        const _Float16* wrrow = &sWr[n * WROW];
        // Batch all LDS fragment loads before the WMMA chain.
        v16h bl0 = load_b_frag(wlrow, 0, hi);
        v16h bl1 = load_b_frag(wlrow, 1, hi);
        v16h br0 = load_b_frag(wrrow, 0, hi);
        v16h br1 = load_b_frag(wrrow, 1, hi);

        v8f c0, c1;
#pragma unroll
        for (int r = 0; r < 8; ++r) { c0[r] = bv; c1[r] = 0.0f; }

        // Two independent accumulator chains -> WMMA pairs can pipeline.
        c0 = __builtin_amdgcn_wmma_f32_16x16x32_f16(false, aAgg0, false, bl0,
                                                    (short)0, c0, false, false);
        c1 = __builtin_amdgcn_wmma_f32_16x16x32_f16(false, aH0, false, br0,
                                                    (short)0, c1, false, false);
        c0 = __builtin_amdgcn_wmma_f32_16x16x32_f16(false, aAgg1, false, bl1,
                                                    (short)0, c0, false, false);
        c1 = __builtin_amdgcn_wmma_f32_16x16x32_f16(false, aH1, false, br1,
                                                    (short)0, c1, false, false);

        if (n < outDim) {
            int mb = hi ? 8 : 0;  // D rows for this lane: mb .. mb+7
#pragma unroll
            for (int r = 0; r < 8; ++r) {
                float v = c0[r] + c1[r];
                if (relu) v = fmaxf(v, 0.0f);
                out[(long long)(nodeBase + mb + r) * outStride + n] = v;
            }
        }
    }
}

// ---------------------------------------------------------------------------
// Row-wise log-softmax in place on [N x C] logits.
// ---------------------------------------------------------------------------
__global__ void sage_logsoftmax_kernel(float* __restrict__ out, int nNodes,
                                       int C) {
    int i = blockIdx.x * blockDim.x + threadIdx.x;
    if (i >= nNodes) return;
    float* row = out + (long long)i * C;
    float mx = -3.4e38f;
    for (int c = 0; c < C; ++c) mx = fmaxf(mx, row[c]);
    float s = 0.0f;
    for (int c = 0; c < C; ++c) s += __expf(row[c] - mx);
    float l = mx + __logf(s);
    for (int c = 0; c < C; ++c) row[c] -= l;
}

extern "C" void kernel_launch(void* const* d_in, const int* in_sizes, int n_in,
                              void* d_out, int out_size, void* d_ws,
                              size_t ws_size, hipStream_t stream) {
    const float* x   = (const float*)d_in[0];
    const int*   ei  = (const int*)d_in[1];
    const float* Wl1 = (const float*)d_in[2];
    const float* Wr1 = (const float*)d_in[3];
    const float* b1  = (const float*)d_in[4];
    const float* Wl2 = (const float*)d_in[5];
    const float* Wr2 = (const float*)d_in[6];
    const float* b2  = (const float*)d_in[7];
    const float* Wl3 = (const float*)d_in[8];
    const float* Wr3 = (const float*)d_in[9];
    const float* b3  = (const float*)d_in[10];

    const int N = in_sizes[0] / F_DIM;      // 100000 (divisible by 16)
    const int E = in_sizes[1] / 2;          // 1600000
    const int H = in_sizes[4];              // 64
    const int C = in_sizes[10];             // 40
    const int Cpad = ((C + 15) / 16) * 16;  // 48

    const int* src = ei;
    const int* dst = ei + E;

    // Workspace: deg[N] | agg[N*64] | h1[N*64] | h2[N*64] | f16 W's | b3 pad
    char*  w    = (char*)d_ws;
    float* deg  = (float*)w;  w += (size_t)N * sizeof(float);
    float* aggb = (float*)w;  w += (size_t)N * F_DIM * sizeof(float);
    float* h1   = (float*)w;  w += (size_t)N * F_DIM * sizeof(float);
    float* h2   = (float*)w;  w += (size_t)N * F_DIM * sizeof(float);
    _Float16* Wl1h = (_Float16*)w;  w += (size_t)H * F_DIM * sizeof(_Float16);
    _Float16* Wr1h = (_Float16*)w;  w += (size_t)H * F_DIM * sizeof(_Float16);
    _Float16* Wl2h = (_Float16*)w;  w += (size_t)H * F_DIM * sizeof(_Float16);
    _Float16* Wr2h = (_Float16*)w;  w += (size_t)H * F_DIM * sizeof(_Float16);
    _Float16* Wl3h = (_Float16*)w;  w += (size_t)Cpad * F_DIM * sizeof(_Float16);
    _Float16* Wr3h = (_Float16*)w;  w += (size_t)Cpad * F_DIM * sizeof(_Float16);
    float* b3p  = (float*)w;  w += (size_t)Cpad * sizeof(float);

    float* logits = (float*)d_out;

    const int BLK = 256;
    // f32 -> f16 weight conversion (zero-padded for layer 3) + bias pad.
    int wtot = H * F_DIM;
    sage_wcvt_kernel<<<(wtot + BLK - 1) / BLK, BLK, 0, stream>>>(Wl1, Wl1h, H, H);
    sage_wcvt_kernel<<<(wtot + BLK - 1) / BLK, BLK, 0, stream>>>(Wr1, Wr1h, H, H);
    sage_wcvt_kernel<<<(wtot + BLK - 1) / BLK, BLK, 0, stream>>>(Wl2, Wl2h, H, H);
    sage_wcvt_kernel<<<(wtot + BLK - 1) / BLK, BLK, 0, stream>>>(Wr2, Wr2h, H, H);
    int wtot3 = Cpad * F_DIM;
    sage_wcvt_kernel<<<(wtot3 + BLK - 1) / BLK, BLK, 0, stream>>>(Wl3, Wl3h, C, Cpad);
    sage_wcvt_kernel<<<(wtot3 + BLK - 1) / BLK, BLK, 0, stream>>>(Wr3, Wr3h, C, Cpad);
    sage_bcvt_kernel<<<1, BLK, 0, stream>>>(b3, b3p, C, Cpad);

    // Degree (once)
    hipMemsetAsync(deg, 0, (size_t)N * sizeof(float), stream);
    sage_degree_kernel<<<(E + BLK - 1) / BLK, BLK, 0, stream>>>(dst, deg, E);

    const int scatterBlocks = (E * 16 + BLK - 1) / BLK;
    const int nWaves = N / 16;  // N % 16 == 0
    const int gemmBlocks = (nWaves * 32 + BLK - 1) / BLK;
    const size_t aggBytes = (size_t)N * F_DIM * sizeof(float);

    // ---- Layer 1: x -> h1 (ReLU) ----
    hipMemsetAsync(aggb, 0, aggBytes, stream);
    sage_scatter_kernel<<<scatterBlocks, BLK, 0, stream>>>(x, src, dst, aggb, E);
    sage_gemm_kernel<4><<<gemmBlocks, BLK, 0, stream>>>(
        aggb, x, deg, Wl1h, Wr1h, b1, h1, H, H, 1, nWaves);

    // ---- Layer 2: h1 -> h2 (ReLU) ----
    hipMemsetAsync(aggb, 0, aggBytes, stream);
    sage_scatter_kernel<<<scatterBlocks, BLK, 0, stream>>>(h1, src, dst, aggb, E);
    sage_gemm_kernel<4><<<gemmBlocks, BLK, 0, stream>>>(
        aggb, h1, deg, Wl2h, Wr2h, b2, h2, H, H, 1, nWaves);

    // ---- Layer 3: h2 -> logits (no ReLU), then log-softmax in place ----
    hipMemsetAsync(aggb, 0, aggBytes, stream);
    sage_scatter_kernel<<<scatterBlocks, BLK, 0, stream>>>(h2, src, dst, aggb, E);
    sage_gemm_kernel<3><<<gemmBlocks, BLK, 0, stream>>>(
        aggb, h2, deg, Wl3h, Wr3h, b3p, logits, C, C, 0, nWaves);

    sage_logsoftmax_kernel<<<(N + BLK - 1) / BLK, BLK, 0, stream>>>(logits, N, C);
}
